// ZoomMIL_25357486915778
// MI455X (gfx1250) — matrix operand order repaired and checked by reading, not compiled, not run on hardware
//
#include <hip/hip_runtime.h>
#include <math.h>

typedef __attribute__((ext_vector_type(16))) _Float16 v16h;
typedef __attribute__((ext_vector_type(2)))  _Float16 h2f;
typedef __attribute__((ext_vector_type(8)))  float    v8f;

#define TILE_COLS 128   // block computes 32 x 128 output tile: 8 waves x 2 (16x16) tiles
#define POOL_CHUNK 256  // rows per block in partial pooling

// ---------------------------------------------------------------------------
// Generic WMMA GEMM: Y[m][n] = act( sum_k X[rowmap[m]][k] * W[k][n] + bias[n] )
// X: (srcRows x K) fp32, W: (K x F) fp32, Y: (N x F) fp32. K % 32 == 0,
// N % 32 == 0, F % TILE_COLS == 0. rowmap may be null (identity).
// Tiles staged into LDS pre-swizzled into the ISA 7.12.2 fragment layout,
// so each lane reads its whole WMMA operand as one aligned v16h (2x b128).
// ---------------------------------------------------------------------------
__global__ __launch_bounds__(256)
void zmil_gemm_wmma(const float* __restrict__ X, const int* __restrict__ rowmap,
                    const float* __restrict__ W, const float* __restrict__ bias,
                    float* __restrict__ Y, int N, int K, int F, int do_relu) {
    __shared__ __align__(16) _Float16 sAf[2 * 32 * 16];   // 2 row-tiles (32 rows x 32 k)
    __shared__ __align__(16) _Float16 sBf[8 * 32 * 16];   // 8 col-tiles (32 k x 128 f)

    const int rowbase = blockIdx.x * 32;
    const int colbase = blockIdx.y * TILE_COLS;
    const int tid  = threadIdx.x;
    const int wave = tid >> 5;
    const int lane = tid & 31;

    v8f acc0 = {}, acc1 = {};

    // ---- A staging setup: this thread loads 4 consecutive k of one row ----
    const int a_m32 = tid >> 3;          // row within 32-row tile
    const int a_k0  = (tid & 7) * 4;     // k start, multiple of 4
    const int a_src = rowmap ? rowmap[rowbase + a_m32] : (rowbase + a_m32);
    const float* Xrow = X + (size_t)a_src * K + a_k0;
    // destination LDS slots (constant): two adjacent h2 pairs -> merged b64 store
    int a_dst[2];
    {
        const int a_r = a_m32 >> 4, a_m = a_m32 & 15;
#pragma unroll
        for (int j = 0; j < 4; j += 2) {
            int k   = a_k0 + j;
            int seg = k >> 3;
            int hi  = seg & 1;
            int v   = (seg >> 1) * 4 + ((k & 7) >> 1);
            a_dst[j >> 1] = ((a_r * 32) + (a_m + 16 * hi)) * 16 + 2 * v;
        }
    }

    // ---- B staging setup: 8 (k-pair, f) slots per thread, constant offsets ----
    int b_off[8];   // element offset of (k0,f) relative to W + colbase
    int b_dst[8];   // LDS destination index
#pragma unroll
    for (int i = 0; i < 8; ++i) {
        int p  = i * 256 + tid;
        int f  = p & 127;
        int k0 = (p >> 7) * 2;
        b_off[i] = k0 * F + f;
        int wv = f >> 4, fcol = f & 15;
        b_dst[i] = ((wv * 32) + (fcol + 16 * ((k0 >> 3) & 1))) * 16 + ((k0 >> 4) * 8 + (k0 & 7));
    }
    const float* Wb = W + colbase;
    size_t kkF = 0;
    const size_t stepF = 32 * (size_t)F;

    for (int kk = 0; kk < K; kk += 32) {
        // ---- stage A (32 rows x 32 k) ----
        {
            const float4 vv = *(const float4*)&Xrow[kk];
            h2f p0; p0[0] = (_Float16)vv.x; p0[1] = (_Float16)vv.y;
            h2f p1; p1[0] = (_Float16)vv.z; p1[1] = (_Float16)vv.w;
            *reinterpret_cast<h2f*>(&sAf[a_dst[0]]) = p0;
            *reinterpret_cast<h2f*>(&sAf[a_dst[1]]) = p1;
        }
        // ---- stage B (32 k x 128 f): issue all 16 loads, then convert+store ----
        float r0[8], r1[8];
#pragma unroll
        for (int i = 0; i < 8; ++i) {
            r0[i] = Wb[kkF + b_off[i]];
            r1[i] = Wb[kkF + b_off[i] + F];
        }
#pragma unroll
        for (int i = 0; i < 8; ++i) {
            h2f pp; pp[0] = (_Float16)r0[i]; pp[1] = (_Float16)r1[i];
            *reinterpret_cast<h2f*>(&sBf[b_dst[i]]) = pp;
        }
        kkF += stepF;
        // prefetch next K-step tiles
        if (kk + 32 < K) {
            __builtin_prefetch(&Xrow[kk + 32], 0, 1);
            __builtin_prefetch(&Wb[kkF + b_off[0]], 0, 1);
        }
        __syncthreads();

        const v16h a0 = *reinterpret_cast<const v16h*>(&sAf[(0 * 32 + lane) * 16]);
        const v16h a1 = *reinterpret_cast<const v16h*>(&sAf[(1 * 32 + lane) * 16]);
        const v16h bb = *reinterpret_cast<const v16h*>(&sBf[(wave * 32 + lane) * 16]);

        acc0 = __builtin_amdgcn_wmma_f32_16x16x32_f16(
            false, a0, false, bb, (short)0, acc0, false, false);
        acc1 = __builtin_amdgcn_wmma_f32_16x16x32_f16(
            false, a1, false, bb, (short)0, acc1, false, false);
        __syncthreads();
    }

    // writeback: VGPR v -> M = v + 8*(lane>=16), N = lane%16
    const int n = colbase + wave * 16 + (lane & 15);
    const float bv = bias ? bias[n] : 0.0f;
    float* Yp = Y + (size_t)(rowbase + ((lane < 16) ? 0 : 8)) * F + n;
#pragma unroll
    for (int v = 0; v < 8; ++v) {
        float val0 = acc0[v] + bv;
        float val1 = acc1[v] + bv;
        if (do_relu) { val0 = fmaxf(val0, 0.0f); val1 = fmaxf(val1, 0.0f); }
        Yp[(size_t)v * F] = val0;
        Yp[(size_t)(v + 16) * F] = val1;
    }
}

// ---------------------------------------------------------------------------
// Gated-attention score: score[n] = sum_h tanh(Ya)*sigmoid(Yg)*Wc[h] + bc
// one block (256 threads) per row; H == 256
// ---------------------------------------------------------------------------
__global__ __launch_bounds__(256)
void zmil_ga_score(const float* __restrict__ Ya, const float* __restrict__ Yg,
                   const float* __restrict__ Wc, const float* __restrict__ bc,
                   float* __restrict__ score, int H) {
    __shared__ float red[256];
    const int n = blockIdx.x;
    const int t = threadIdx.x;
    float v = 0.0f;
    if (t < H) {
        float a = tanhf(Ya[(size_t)n * H + t]);
        float g = 1.0f / (1.0f + expf(-Yg[(size_t)n * H + t]));
        v = a * g * Wc[t];
    }
    red[t] = v;
    __syncthreads();
    for (int s = 128; s > 0; s >>= 1) {
        if (t < s) red[t] += red[t + s];
        __syncthreads();
    }
    if (t == 0) score[n] = red[0] + bc[0];
}

// ---------------------------------------------------------------------------
// wout[n] = softmax(score)[n]; single block, 512 threads, N <= 4096
// ---------------------------------------------------------------------------
__global__ __launch_bounds__(512)
void zmil_softmax_weights(const float* __restrict__ score, float* __restrict__ wout,
                          int N) {
    __shared__ float red[512];
    const int t = threadIdx.x;

    float mx = -3.0e38f;
    for (int i = t; i < N; i += 512) mx = fmaxf(mx, score[i]);
    red[t] = mx;
    __syncthreads();
    for (int s = 256; s > 0; s >>= 1) {
        if (t < s) red[t] = fmaxf(red[t], red[t + s]);
        __syncthreads();
    }
    mx = red[0];
    __syncthreads();

    float z = 0.0f;
    for (int i = t; i < N; i += 512) {
        float ev = expf(score[i] - mx);
        wout[i] = ev;
        z += ev;
    }
    red[t] = z;
    __syncthreads();
    for (int s = 256; s > 0; s >>= 1) {
        if (t < s) red[t] += red[t + s];
        __syncthreads();
    }
    const float inv = 1.0f / red[0];
    __syncthreads();
    for (int i = t; i < N; i += 512) wout[i] *= inv;
}

// partial[b][f] = sum over POOL_CHUNK rows of w[n]*Hm[n][f]; 512 threads (f), F=512
__global__ __launch_bounds__(512)
void zmil_wpool_partial(const float* __restrict__ w, const float* __restrict__ Hm,
                        float* __restrict__ partial, int F) {
    const int t = threadIdx.x;
    const int base = blockIdx.x * POOL_CHUNK;
    const float* hp = Hm + (size_t)base * F + t;
    const float* wp = w + base;
    float acc = 0.0f;
#pragma unroll 4
    for (int n = 0; n < POOL_CHUNK; ++n) {
        acc += wp[n] * hp[0];
        hp += F;
    }
    partial[(size_t)blockIdx.x * F + t] = acc;
}

// M[f] = sum_b partial[b][f]; 512 threads
__global__ __launch_bounds__(512)
void zmil_wpool_reduce(const float* __restrict__ partial, float* __restrict__ M,
                       int nblocks, int F) {
    const int t = threadIdx.x;
    const float* pp = partial + t;
    float acc = 0.0f;
    for (int b = 0; b < nblocks; ++b) {
        acc += pp[0];
        pp += F;
    }
    M[t] = acc;
}

// ---------------------------------------------------------------------------
// top-k by rank (matches lax.top_k tie-break: equal values -> lower index wins),
// output indices emitted in ascending index order (== jnp.sort of top_k idx).
// single block, 1024 threads, N <= 4096.
// ---------------------------------------------------------------------------
__global__ __launch_bounds__(1024)
void zmil_topk_sorted(const float* __restrict__ score, int* __restrict__ idx,
                      int N, int k) {
    __shared__ float sc[4096];
    __shared__ unsigned char flag[4096];
    const int t = threadIdx.x;
    for (int i = t; i < N; i += 1024) sc[i] = score[i];
    __syncthreads();
    for (int i = t; i < N; i += 1024) {
        float si = sc[i];
        int rank = 0;
        for (int j = 0; j < N; ++j) {
            float sj = sc[j];
            rank += (sj > si) || (sj == si && j < i);
        }
        flag[i] = (rank < k) ? 1 : 0;
    }
    __syncthreads();
    if (t == 0) {
        int p = 0;
        for (int i = 0; i < N; ++i)
            if (flag[i]) idx[p++] = i;
    }
}

// rowmap2[r] = idx1[r/ratio]*ratio + r%ratio
__global__ void zmil_rowmap_blocks(const int* __restrict__ idx, int* __restrict__ rowmap,
                                   int total, int ratio) {
    int r = blockIdx.x * blockDim.x + threadIdx.x;
    if (r < total) rowmap[r] = idx[r / ratio] * ratio + (r % ratio);
}

// rowmap3[r]: t = idx2[r/4]*4 + r%4 (row of x3s); row of x3 = idx1[t/16]*16 + t%16
__global__ void zmil_rowmap_compose(const int* __restrict__ idx1, const int* __restrict__ idx2,
                                    int* __restrict__ rowmap, int total) {
    int r = blockIdx.x * blockDim.x + threadIdx.x;
    if (r < total) {
        int t = idx2[r / 4] * 4 + (r % 4);
        rowmap[r] = idx1[t / 16] * 16 + (t % 16);
    }
}

// logits[c] = bh[c] + sum_f (M1+M2+M3)[f] * Wh[f][c]; 512 threads, F=512, C=4
__global__ __launch_bounds__(512)
void zmil_head(const float* __restrict__ M1, const float* __restrict__ M2,
               const float* __restrict__ M3, const float* __restrict__ Wh,
               const float* __restrict__ bh, float* __restrict__ out, int C) {
    __shared__ float red[512];
    const int t = threadIdx.x;
    const float m = M1[t] + M2[t] + M3[t];
    for (int c = 0; c < C; ++c) {
        red[t] = m * Wh[(size_t)t * C + c];
        __syncthreads();
        for (int s = 256; s > 0; s >>= 1) {
            if (t < s) red[t] += red[t + s];
            __syncthreads();
        }
        if (t == 0) out[c] = red[0] + bh[c];
        __syncthreads();
    }
}

// ---------------------------------------------------------------------------
extern "C" void kernel_launch(void* const* d_in, const int* in_sizes, int n_in,
                              void* d_out, int out_size, void* d_ws, size_t ws_size,
                              hipStream_t stream) {
    const int N1 = 4096, D = 1024, F = 512, H = 256, NCLS = 4, K = 256;
    const int NSUB = K * 4; // 1024 gathered rows at levels 2 and 3

    const float* x1   = (const float*)d_in[0];
    const float* x2   = (const float*)d_in[1];
    const float* x3   = (const float*)d_in[2];
    const float* fcW  = (const float*)d_in[3];   // (3, D, F)
    const float* fcb  = (const float*)d_in[4];   // (3, F)
    const float* gaWa = (const float*)d_in[5];   // (5, F, H)
    const float* gaba = (const float*)d_in[6];   // (5, H)
    const float* gaWb = (const float*)d_in[7];   // (5, F, H)
    const float* gabb = (const float*)d_in[8];   // (5, H)
    const float* gaWc = (const float*)d_in[9];   // (5, H, 1)
    const float* gabc = (const float*)d_in[10];  // (5, 1)
    const float* Wh   = (const float*)d_in[11];  // (F, NCLS)
    const float* bh   = (const float*)d_in[12];  // (NCLS,)
    float* out = (float*)d_out;

    // workspace layout
    float* ws = (float*)d_ws;
    float* h1 = ws;                       // 4096*512
    float* h2 = h1 + (size_t)N1 * F;      // 1024*512
    float* h3 = h2 + (size_t)NSUB * F;    // 1024*512
    float* Ya = h3 + (size_t)NSUB * F;    // 4096*256
    float* Yg = Ya + (size_t)N1 * H;      // 4096*256
    float* s_main = Yg + (size_t)N1 * H;  // 4096
    float* s_aux  = s_main + N1;          // 4096
    float* wnorm  = s_aux + N1;           // 4096
    float* partial = wnorm + N1;          // 16*512
    float* M1 = partial + 16 * F;         // 512
    float* M2 = M1 + F;
    float* M3 = M2 + F;
    int* idx1    = (int*)(M3 + F);        // 256
    int* idx2    = idx1 + K;              // 256
    int* rowmap2 = idx2 + K;              // 1024
    int* rowmap3 = rowmap2 + NSUB;        // 1024

    dim3 blk(256);
    const int PB1 = N1 / POOL_CHUNK;      // 16 pooling blocks at level 1
    const int PBS = NSUB / POOL_CHUNK;    // 4 at levels 2/3

    // ---------------- level 1 (low mag, 4096 rows) ----------------
    zmil_gemm_wmma<<<dim3(N1 / 32, F / TILE_COLS), blk, 0, stream>>>(
        x1, nullptr, fcW + 0 * (size_t)D * F, fcb + 0 * F, h1, N1, D, F, 1);

    zmil_gemm_wmma<<<dim3(N1 / 32, H / TILE_COLS), blk, 0, stream>>>(
        h1, nullptr, gaWa + 0 * (size_t)F * H, gaba + 0 * H, Ya, N1, F, H, 0);
    zmil_gemm_wmma<<<dim3(N1 / 32, H / TILE_COLS), blk, 0, stream>>>(
        h1, nullptr, gaWb + 0 * (size_t)F * H, gabb + 0 * H, Yg, N1, F, H, 0);
    zmil_ga_score<<<N1, 256, 0, stream>>>(Ya, Yg, gaWc + 0 * H, gabc + 0, s_main, H);
    zmil_softmax_weights<<<1, 512, 0, stream>>>(s_main, wnorm, N1);
    zmil_wpool_partial<<<PB1, 512, 0, stream>>>(wnorm, h1, partial, F);
    zmil_wpool_reduce<<<1, 512, 0, stream>>>(partial, M1, PB1, F);

    zmil_gemm_wmma<<<dim3(N1 / 32, H / TILE_COLS), blk, 0, stream>>>(
        h1, nullptr, gaWa + 3 * (size_t)F * H, gaba + 3 * H, Ya, N1, F, H, 0);
    zmil_gemm_wmma<<<dim3(N1 / 32, H / TILE_COLS), blk, 0, stream>>>(
        h1, nullptr, gaWb + 3 * (size_t)F * H, gabb + 3 * H, Yg, N1, F, H, 0);
    zmil_ga_score<<<N1, 256, 0, stream>>>(Ya, Yg, gaWc + 3 * H, gabc + 3, s_aux, H);
    zmil_topk_sorted<<<1, 1024, 0, stream>>>(s_aux, idx1, N1, K);
    zmil_rowmap_blocks<<<(NSUB + 255) / 256, 256, 0, stream>>>(idx1, rowmap2, NSUB, 4);

    // ---------------- level 2 (mid mag, 1024 gathered rows) ----------------
    zmil_gemm_wmma<<<dim3(NSUB / 32, F / TILE_COLS), blk, 0, stream>>>(
        x2, rowmap2, fcW + 1 * (size_t)D * F, fcb + 1 * F, h2, NSUB, D, F, 1);

    zmil_gemm_wmma<<<dim3(NSUB / 32, H / TILE_COLS), blk, 0, stream>>>(
        h2, nullptr, gaWa + 1 * (size_t)F * H, gaba + 1 * H, Ya, NSUB, F, H, 0);
    zmil_gemm_wmma<<<dim3(NSUB / 32, H / TILE_COLS), blk, 0, stream>>>(
        h2, nullptr, gaWb + 1 * (size_t)F * H, gabb + 1 * H, Yg, NSUB, F, H, 0);
    zmil_ga_score<<<NSUB, 256, 0, stream>>>(Ya, Yg, gaWc + 1 * H, gabc + 1, s_main, H);
    zmil_softmax_weights<<<1, 512, 0, stream>>>(s_main, wnorm, NSUB);
    zmil_wpool_partial<<<PBS, 512, 0, stream>>>(wnorm, h2, partial, F);
    zmil_wpool_reduce<<<1, 512, 0, stream>>>(partial, M2, PBS, F);

    zmil_gemm_wmma<<<dim3(NSUB / 32, H / TILE_COLS), blk, 0, stream>>>(
        h2, nullptr, gaWa + 4 * (size_t)F * H, gaba + 4 * H, Ya, NSUB, F, H, 0);
    zmil_gemm_wmma<<<dim3(NSUB / 32, H / TILE_COLS), blk, 0, stream>>>(
        h2, nullptr, gaWb + 4 * (size_t)F * H, gabb + 4 * H, Yg, NSUB, F, H, 0);
    zmil_ga_score<<<NSUB, 256, 0, stream>>>(Ya, Yg, gaWc + 4 * H, gabc + 4, s_aux, H);
    zmil_topk_sorted<<<1, 1024, 0, stream>>>(s_aux, idx2, NSUB, K);
    zmil_rowmap_compose<<<(NSUB + 255) / 256, 256, 0, stream>>>(idx1, idx2, rowmap3, NSUB);

    // ---------------- level 3 (high mag, 1024 gathered rows) ----------------
    zmil_gemm_wmma<<<dim3(NSUB / 32, F / TILE_COLS), blk, 0, stream>>>(
        x3, rowmap3, fcW + 2 * (size_t)D * F, fcb + 2 * F, h3, NSUB, D, F, 1);

    zmil_gemm_wmma<<<dim3(NSUB / 32, H / TILE_COLS), blk, 0, stream>>>(
        h3, nullptr, gaWa + 2 * (size_t)F * H, gaba + 2 * H, Ya, NSUB, F, H, 0);
    zmil_gemm_wmma<<<dim3(NSUB / 32, H / TILE_COLS), blk, 0, stream>>>(
        h3, nullptr, gaWb + 2 * (size_t)F * H, gabb + 2 * H, Yg, NSUB, F, H, 0);
    zmil_ga_score<<<NSUB, 256, 0, stream>>>(Ya, Yg, gaWc + 2 * H, gabc + 2, s_main, H);
    zmil_softmax_weights<<<1, 512, 0, stream>>>(s_main, wnorm, NSUB);
    zmil_wpool_partial<<<PBS, 512, 0, stream>>>(wnorm, h3, partial, F);
    zmil_wpool_reduce<<<1, 512, 0, stream>>>(partial, M3, PBS, F);

    // ---------------- head ----------------
    zmil_head<<<1, 512, 0, stream>>>(M1, M2, M3, Wh, bh, out, NCLS);

    (void)in_sizes; (void)n_in; (void)out_size; (void)ws_size;
}
